// ConvNACCell_5411658793293
// MI455X (gfx1250) — compile-verified, matrix operand order checked by reading, not run on hardware
//
#include <hip/hip_runtime.h>

typedef __attribute__((ext_vector_type(16))) __bf16 v16bf;
typedef __attribute__((ext_vector_type(8)))  __bf16 v8bf;
typedef __attribute__((ext_vector_type(4)))  __bf16 v4bf;
typedef __attribute__((ext_vector_type(8)))  float  v8f;

#define CH_IN   64
#define CH_OUT  128
#define HW      256
#define KTOT    576            // CH_IN * 3 * 3
#define CP      72             // padded channel stride in LDS (mult of 8 -> 16B-aligned b128)
#define WTILE   128            // spatial positions per block (8 N-tiles of 16)
#define WEXT    (WTILE + 2)    // halo width

// ---------------------------------------------------------------------------
// Stage 1: NAC weight transform + K-reorder + bf16 pack.
//   W = tanh(W_hat) * sigmoid(M_hat), original k = c*9 + (kh*3+kw)
//   stored as Wb[cout][k'] with k' = (kh*3+kw)*64 + c  (tap-major, channel-contig)
// ---------------------------------------------------------------------------
__global__ void nac_weight_transform(const float* __restrict__ W_hat,
                                     const float* __restrict__ M_hat,
                                     __bf16* __restrict__ Wb) {
    int i = blockIdx.x * blockDim.x + threadIdx.x;
    if (i >= CH_OUT * KTOT) return;
    int cout = i / KTOT;
    int k    = i - cout * KTOT;
    int c    = k / 9;
    int tap  = k - c * 9;
    float w  = tanhf(W_hat[i]);
    float m  = 1.0f / (1.0f + __expf(-M_hat[i]));
    int kp   = tap * CH_IN + c;
    Wb[cout * KTOT + kp] = (__bf16)(w * m);
}

// ---------------------------------------------------------------------------
// Stage 2: implicit-GEMM 3x3 conv via v_wmma_f32_16x16x32_bf16.
// Block = 256 thr (8 wave32). Block tile: 128 spatial (row strip) x 128 cout.
// Wave w: cout pair (wave&3 -> couts [32p,32p+32)) x spatial half (wave>>2,
// 4 N-tiles of 16). Per K-chunk: 2 A fragments (4 global b128, reused 4x) +
// 4 B fragments (8 ds b128, reused 2x) + 8 WMMA -> 1.5 mem-ops per WMMA.
// ---------------------------------------------------------------------------
__global__ __launch_bounds__(256)
void conv_nac_wmma(const float* __restrict__ x,
                   const __bf16* __restrict__ Wb,
                   float* __restrict__ out) {
    // x halo tile in LDS, layout [kh][w'][c] bf16, w' in [0,WEXT), c padded to CP
    __shared__ __bf16 tile[3 * WEXT * CP];

    const int bid = blockIdx.x;           // 8192 blocks: [n][h][wstrip]
    const int wt  = bid & 1;              // 2 strips of 128 per row
    const int h   = (bid >> 1) & 255;
    const int n   = bid >> 9;
    const int w0  = wt * WTILE;
    const int tid = threadIdx.x;

    // ---- stage x tile (fp32 -> bf16), zero-padded halo -------------------
    // Each iteration: 4 consecutive channels, one (hh,ww) -> one ds_store_b64.
    for (int i = tid; i < 3 * (CH_IN / 4) * WEXT; i += 256) {
        int wp   = i % WEXT;              // w' innermost: coalesced global reads
        int rest = i / WEXT;
        int cq   = rest & 15;             // channel quad
        int kh   = rest >> 4;
        int hh   = h + kh - 1;
        int ww   = w0 + wp - 1;
        int c    = cq * 4;
        v4bf v   = {};
        if ((unsigned)hh < (unsigned)HW && (unsigned)ww < (unsigned)HW) {
            const float* xp = x + (((size_t)n * CH_IN + c) * HW + hh) * HW + ww;
            v[0] = (__bf16)xp[0 * HW * HW];
            v[1] = (__bf16)xp[1 * HW * HW];
            v[2] = (__bf16)xp[2 * HW * HW];
            v[3] = (__bf16)xp[3 * HW * HW];
        }
        *(v4bf*)&tile[(kh * WEXT + wp) * CP + c] = v;
    }
    __syncthreads();

    // ---- per-wave WMMA ----------------------------------------------------
    const int wave  = tid >> 5;
    const int lane  = tid & 31;
    const int col   = lane & 15;          // = M row for A, = N col for B
    const int khalf = lane >> 4;          // K-half select per 16-bit A/B layout
    const int cpair = wave & 3;           // cout pair -> couts [32*cpair, +32)
    const int sgrp  = wave >> 2;          // spatial half -> offset sgrp*64
    const int coutBase = cpair * 32;

    const __bf16* arow0 = Wb + (size_t)(coutBase + col) * KTOT + khalf * 8;
    const __bf16* arow1 = arow0 + (size_t)16 * KTOT;

    v8f acc[8];
#pragma unroll
    for (int q = 0; q < 8; ++q) acc[q] = (v8f){};

#pragma unroll
    for (int kc = 0; kc < 18; ++kc) {     // K chunks of 32: tap = kc/2, c0 = (kc&1)*32
        const int tap = kc >> 1;
        const int kh  = tap / 3;
        const int kw  = tap - kh * 3;
        const int c0  = (kc & 1) * 32;

        // Two A fragments (cout tiles), 2 x 16B global loads each (L2-resident)
        const __bf16* ap0 = arow0 + kc * 32;
        const __bf16* ap1 = arow1 + kc * 32;
        v16bf a0 = __builtin_shufflevector(*(const v8bf*)ap0, *(const v8bf*)(ap0 + 16),
                     0,1,2,3,4,5,6,7, 8,9,10,11,12,13,14,15);
        v16bf a1 = __builtin_shufflevector(*(const v8bf*)ap1, *(const v8bf*)(ap1 + 16),
                     0,1,2,3,4,5,6,7, 8,9,10,11,12,13,14,15);

        const int lrow = kh * WEXT + sgrp * 64 + col + kw;
#pragma unroll
        for (int s = 0; s < 4; ++s) {     // 4 spatial N-tiles
            const __bf16* bp = &tile[(lrow + s * 16) * CP + c0 + khalf * 8];
            v16bf b = __builtin_shufflevector(*(const v8bf*)bp, *(const v8bf*)(bp + 16),
                        0,1,2,3,4,5,6,7, 8,9,10,11,12,13,14,15);
            acc[2*s+0] = __builtin_amdgcn_wmma_f32_16x16x32_bf16(
                             false, a0, false, b, (short)0, acc[2*s+0], false, false);
            acc[2*s+1] = __builtin_amdgcn_wmma_f32_16x16x32_bf16(
                             false, a1, false, b, (short)0, acc[2*s+1], false, false);
        }
    }

    // ---- store D: VGPR r -> cout = base + khalf*8 + r, N = col (coalesced) --
    size_t obase = (((size_t)n * CH_OUT + coutBase + khalf * 8) * HW + h) * HW
                 + w0 + sgrp * 64 + col;
#pragma unroll
    for (int r = 0; r < 8; ++r) {
        size_t o0 = obase + (size_t)r * (HW * HW);            // cout tile 0
        size_t o1 = o0 + (size_t)16 * (HW * HW);              // cout tile 1
#pragma unroll
        for (int s = 0; s < 4; ++s) {
            out[o0 + s * 16] = acc[2*s+0][r];
            out[o1 + s * 16] = acc[2*s+1][r];
        }
    }
}

// ---------------------------------------------------------------------------
extern "C" void kernel_launch(void* const* d_in, const int* in_sizes, int n_in,
                              void* d_out, int out_size, void* d_ws, size_t ws_size,
                              hipStream_t stream) {
    const float* x     = (const float*)d_in[0];
    const float* W_hat = (const float*)d_in[1];
    const float* M_hat = (const float*)d_in[2];
    __bf16* Wb   = (__bf16*)d_ws;                    // 128*576*2 = 147456 B scratch
    float*  out  = (float*)d_out;

    nac_weight_transform<<<(CH_OUT * KTOT + 255) / 256, 256, 0, stream>>>(
        W_hat, M_hat, Wb);

    const int nblocks = 16 * HW * (HW / WTILE);       // n * h * wstrips = 8192
    conv_nac_wmma<<<nblocks, 256, 0, stream>>>(x, Wb, out);
}
